// RelativeMultiHeadAttention_83562883711697
// MI455X (gfx1250) — matrix operand order repaired and checked by reading, not compiled, not run on hardware
//
#include <hip/hip_runtime.h>
#include <hip/hip_bf16.h>
#include <math.h>
#include <type_traits>

// ---------- types ----------
typedef __attribute__((ext_vector_type(16))) __bf16 v16bf;
typedef __attribute__((ext_vector_type(8)))  __bf16 v8bf;
typedef __attribute__((ext_vector_type(2)))  __bf16 v2bf;
typedef __attribute__((ext_vector_type(8)))  float  v8f;
typedef __attribute__((ext_vector_type(4)))  float  v4f;
typedef __attribute__((ext_vector_type(4)))  unsigned int u32x4;
typedef __attribute__((ext_vector_type(8)))  int          i32x8;
typedef __attribute__((ext_vector_type(4)))  int          i32x4;

__device__ __forceinline__ v8f wmma_bf16(v16bf a, v16bf b, v8f c) {
  return __builtin_amdgcn_wmma_f32_16x16x32_bf16(false, a, false, b, (short)0, c, false, false);
}

__device__ __forceinline__ v16bf cat16(v8bf lo, v8bf hi) {
  return __builtin_shufflevector(lo, hi, 0,1,2,3,4,5,6,7,8,9,10,11,12,13,14,15);
}

// A-matrix per-lane vector: lane L holds row M=L%16; elems j:
// K = j + 16*(L>>4)  => one contiguous 16-elem run (two 8-elem chunks).
__device__ __forceinline__ v16bf load_a_chunk(const __bf16* row, unsigned hl) {
  v8bf lo = *(const v8bf*)(row + 16u * hl);
  v8bf hi = *(const v8bf*)(row + 16u * hl + 8u);
  return cat16(lo, hi);
}

// ---------------------------------------------------------------------------
// Tensor Data Mover: 2D bf16 tile global -> LDS (D# per ISA cdna5 §8).
// ---------------------------------------------------------------------------
__device__ __forceinline__ void tdm_load_2d_bf16(
    const __bf16* gsrc, const __bf16* ldst,
    unsigned tile_x, unsigned tile_y,
    unsigned tensor_x, unsigned tensor_y, unsigned row_stride)
{
  unsigned long long ga  = (unsigned long long)(const void*)gsrc;
  unsigned           lds = (unsigned)(unsigned long long)(const void*)ldst;
  u32x4 g0 = {};
  g0[0] = 1u;                                            // count=1, user D#
  g0[1] = lds;                                           // lds_addr (bytes)
  g0[2] = (unsigned)ga;                                  // global_addr[31:0]
  g0[3] = (unsigned)((ga >> 32) & 0x01FFFFFFull) | (2u << 30);  // addr[56:32], type=2
  i32x8 g1 = {};
  g1[0] = (int)(1u << 16);                               // data_size=1 -> 2 bytes
  g1[1] = (int)(tensor_x << 16);                         // tensor_dim0 lo16 @bit48
  g1[2] = (int)((tensor_x >> 16) | (tensor_y << 16));    // dim0 hi16 | dim1 lo16
  g1[3] = (int)((tensor_y >> 16) | (tile_x  << 16));     // dim1 hi16 | tile_dim0
  g1[4] = (int)tile_y;                                   // tile_dim1 (tile_dim2=0)
  g1[5] = (int)row_stride;                               // tensor_dim0_stride lo32
  g1[6] = 0; g1[7] = 0;
  i32x4 z4 = {};
#if __clang_major__ >= 23
  i32x8 z8 = {};
  __builtin_amdgcn_tensor_load_to_lds(g0, g1, z4, z4, z8, 0);
#else
  __builtin_amdgcn_tensor_load_to_lds(g0, g1, z4, z4, 0);
#endif
}

// ---------------------------------------------------------------------------
// GEMM: C[M,N] = A[M,K] @ W[K,N] (+ bias[N]).  bf16 WMMA.
// ---------------------------------------------------------------------------
#define GM_TM 64
#define GM_TN 128
#define GM_TK 32

template <typename InT, typename OutT>
__global__ __launch_bounds__(256) void gemm_bias_kernel(
    const InT* __restrict__ A, const float* __restrict__ W,
    const float* __restrict__ bias, OutT* __restrict__ C,
    int M, int N, int K)
{
  __shared__ __bf16 As[GM_TM][GM_TK];   // [m][k]
  __shared__ __bf16 Bs[GM_TN][GM_TK];   // [n][k]

  const int tid = threadIdx.x;
  const unsigned lane = tid & 31u;
  const unsigned wave = (unsigned)tid >> 5;
  const unsigned hl = lane >> 4;
  const unsigned ln = lane & 15u;
  const int m0 = blockIdx.x * GM_TM;
  const int n0 = blockIdx.y * GM_TN;
  const int wm = (int)(wave >> 1);
  const int wn = (int)(wave & 1u);

  v8f acc[4] = {};

  const int arow = tid >> 2;            // 64 rows, 4 thr/row, 8 k each
  const int akk  = (tid & 3) * 8;
  const int bk2  = (tid >> 4) * 2;      // 16 k-pairs
  const int bnn  = (tid & 15) * 8;      // 8 n per thread

  for (int k0 = 0; k0 < K; k0 += GM_TK) {
    __syncthreads();
    { // stage A (row-major)
      const InT* src = A + (size_t)(m0 + arow) * K + k0 + akk;
      if constexpr (std::is_same_v<InT, float>) {
        v4f a0 = *(const v4f*)src;
        v4f a1 = *(const v4f*)(src + 4);
        __bf16* dst = &As[arow][akk];
        #pragma unroll
        for (int j = 0; j < 4; ++j) { dst[j] = (__bf16)a0[j]; dst[4 + j] = (__bf16)a1[j]; }
      } else {
        *(v8bf*)&As[arow][akk] = *(const v8bf*)src;   // already bf16
      }
      if (k0 + GM_TK < K) __builtin_prefetch(src + GM_TK, 0, 1);
    }
    { // stage B transposed to [n][k]; k-pairs packed -> ds_store_b32
      const float* s0 = W + (size_t)(k0 + bk2) * N + n0 + bnn;
      const float* s1 = s0 + N;
      v4f w0a = *(const v4f*)s0,       w1a = *(const v4f*)s1;
      v4f w0b = *(const v4f*)(s0 + 4), w1b = *(const v4f*)(s1 + 4);
      #pragma unroll
      for (int j = 0; j < 4; ++j) {
        v2bf pa; pa[0] = (__bf16)w0a[j]; pa[1] = (__bf16)w1a[j];
        *(v2bf*)&Bs[bnn + j][bk2] = pa;
        v2bf pb; pb[0] = (__bf16)w0b[j]; pb[1] = (__bf16)w1b[j];
        *(v2bf*)&Bs[bnn + 4 + j][bk2] = pb;
      }
    }
    __syncthreads();

    v16bf av = load_a_chunk(&As[wm * 16 + ln][0], hl);
    #pragma unroll
    for (int i = 0; i < 4; ++i) {
      v16bf bv = *(const v16bf*)&Bs[wn * 64 + i * 16 + ln][16u * hl];
      acc[i] = wmma_bf16(av, bv, acc[i]);
    }
  }

  #pragma unroll
  for (int i = 0; i < 4; ++i) {
    int n = n0 + wn * 64 + i * 16 + (int)ln;
    float bval = bias ? bias[n] : 0.0f;
    #pragma unroll
    for (int r = 0; r < 8; ++r) {
      int m = m0 + wm * 16 + r + 8 * (int)hl;
      float val = acc[i][r] + bval;
      if constexpr (std::is_same_v<OutT, float>) {
        __builtin_nontemporal_store(val, &C[(size_t)m * N + n]);
      } else {
        C[(size_t)m * N + n] = (__bf16)val;
      }
    }
  }
}

// ---------------------------------------------------------------------------
// Relative attention.  One workgroup = one (b,h), 32 t-rows.
// 8 waves: wm=wave>>2 (16 rows each), wn=wave&3 (l-quarter of 256 each).
// Scores 16x256/wave = 16 v8f accumulators (fits in 256 VGPRs, no spills).
// K/P tiles DMA'd by the Tensor Data Mover; V staged transposed (b32 pairs).
// ---------------------------------------------------------------------------
#define AT_T  1024
#define AT_DM 512

__global__ __launch_bounds__(256) void attention_kernel(
    const __bf16* __restrict__ Q, const __bf16* __restrict__ Kx,
    const __bf16* __restrict__ Vx, const __bf16* __restrict__ P,
    const float* __restrict__ ubias, const float* __restrict__ vbias,
    float* __restrict__ probs, __bf16* __restrict__ ctx)
{
  extern __shared__ char smem[];
  __bf16 (*kTile)[64][64]     = (__bf16(*)[64][64])(smem);               // 32 KB [4][l][d]
  __bf16 (*pTile)[64][64]     = (__bf16(*)[64][64])(smem + 32*1024);     // 32 KB
  __bf16 (*vT)[64][256]       = (__bf16(*)[64][256])(smem + 64*1024);    // 128 KB [4][d][l]
  __bf16 (*probsLds)[16][256] = (__bf16(*)[16][256])(smem + 192*1024);   // 64 KB [wave][m][l]
  float* redMax = (float*)(smem + 256*1024);                             // [4][32]
  float* redSum = redMax + 128;                                          // [4][32]
  float (*ctxPart)[3][16][64] = (float(*)[3][16][64])(smem);             // 24 KB, reuse kTile

  const int tid = threadIdx.x;
  const unsigned lane = tid & 31u;
  const unsigned wave = (unsigned)tid >> 5;
  const unsigned hl = lane >> 4;
  const unsigned ln = lane & 15u;
  const int wm = (int)(wave >> 2);      // 0..1  (row group)
  const int wn = (int)(wave & 3u);      // 0..3  (l quarter)

  const int bh = blockIdx.y;
  const int b  = bh >> 3;
  const int hd = bh & 7;
  const int t0 = blockIdx.x * 32;
  const int rowt = t0 + wm * 16 + (int)ln;

  // ---- Phase 0: per-lane (q+u) and (q+v) A-vectors ------------------------
  v16bf quv[2], qvv[2];
  {
    const __bf16* qp = Q + ((size_t)(b * AT_T + rowt) * AT_DM) + hd * 64;
    const float* up = ubias + hd * 64;
    const float* vp = vbias + hd * 64;
    #pragma unroll
    for (int s = 0; s < 2; ++s) {
      v8bf c0 = *(const v8bf*)(qp + 32 * s + 16 * (int)hl);
      v8bf c1 = *(const v8bf*)(qp + 32 * s + 16 * (int)hl + 8);
      #pragma unroll
      for (int j = 0; j < 8; ++j) {
        int d0 = 32 * s + 16 * (int)hl + j;
        int d1 = d0 + 8;
        float q0 = (float)c0[j], q1 = (float)c1[j];
        quv[s][j]     = (__bf16)(q0 + up[d0]);
        quv[s][j + 8] = (__bf16)(q1 + up[d1]);
        qvv[s][j]     = (__bf16)(q0 + vp[d0]);
        qvv[s][j + 8] = (__bf16)(q1 + vp[d1]);
      }
    }
  }

  v8f sacc[16] = {};   // 16 rows x 256 cols of raw scores

  // staging within the wn-group (2 waves = 64 threads): 2 l-rows x 32 d each
  const int gtid = (int)((wave >> 2) * 32 + lane);  // 0..63
  const int l2  = (gtid & 31) * 2;                  // even l-row pair
  const int dh  = (gtid >> 5) * 32;                 // d-half

  // ---- Phase 1: scores = (q+u)Kᵀ + (q+v)Pᵀ, 64-l chunks -------------------
  for (int lc = 0; lc < 4; ++lc) {
    __syncthreads();   // prev chunk's consumers done before TDM overwrites
    if (wave == 0) {   // TDM: one descriptor per 64x64 bf16 tile, all quarters
      #pragma unroll
      for (int g = 0; g < 4; ++g) {
        const int lb = g * 256 + lc * 64;
        tdm_load_2d_bf16(Kx + ((size_t)(b * AT_T + lb) * AT_DM) + hd * 64,
                         &kTile[g][0][0], 64, 64, 64, AT_T, AT_DM);
        tdm_load_2d_bf16(P + ((size_t)lb * AT_DM) + hd * 64,
                         &pTile[g][0][0], 64, 64, 64, AT_T, AT_DM);
      }
      __builtin_amdgcn_s_wait_tensorcnt(0);
    }
    { // V staged transposed [d][l]: l-pairs packed -> ds_store_b32
      const int lbase = wn * 256 + lc * 64;
      const __bf16* v0 = Vx + ((size_t)(b * AT_T + lbase + l2) * AT_DM) + hd * 64 + dh;
      const __bf16* v1 = v0 + AT_DM;
      if (lc < 3) __builtin_prefetch(v0 + (size_t)64 * AT_DM, 0, 1);
      #pragma unroll
      for (int c = 0; c < 4; ++c) {
        v8bf a0 = *(const v8bf*)(v0 + c * 8);
        v8bf a1 = *(const v8bf*)(v1 + c * 8);
        #pragma unroll
        for (int j = 0; j < 8; ++j) {
          v2bf pr; pr[0] = a0[j]; pr[1] = a1[j];
          *(v2bf*)&vT[wn][dh + c * 8 + j][lc * 64 + l2] = pr;
        }
      }
    }
    __syncthreads();
    #pragma unroll
    for (int nt = 0; nt < 4; ++nt) {
      const int nl = nt * 16 + (int)ln;
      v8f a = sacc[lc * 4 + nt];
      #pragma unroll
      for (int s = 0; s < 2; ++s) {
        v16bf bk = *(const v16bf*)&kTile[wn][nl][32 * s + 16 * (int)hl];
        a = wmma_bf16(quv[s], bk, a);
        v16bf bp = *(const v16bf*)&pTile[wn][nl][32 * s + 16 * (int)hl];
        a = wmma_bf16(qvv[s], bp, a);
      }
      sacc[lc * 4 + nt] = a;
    }
  }

  // ---- Phase 2: softmax over l (elem r -> row r + 8*hl) -------------------
  const float scale = 0.125f;   // 1/sqrt(64)
  const int ridx = wm * 16 + 8 * (int)hl;   // base row of this lane's stripe
  float rmax[8], rsum[8], rinv[8];
  #pragma unroll
  for (int r = 0; r < 8; ++r) {
    float mx = -3.4e38f;
    #pragma unroll
    for (int i = 0; i < 16; ++i) mx = fmaxf(mx, sacc[i][r]);
    for (int msk = 1; msk < 16; msk <<= 1) mx = fmaxf(mx, __shfl_xor(mx, msk, 32));
    rmax[r] = mx;
  }
  if (ln == 0) {
    #pragma unroll
    for (int r = 0; r < 8; ++r) redMax[wn * 32 + ridx + r] = rmax[r];
  }
  __syncthreads();
  #pragma unroll
  for (int r = 0; r < 8; ++r)
    #pragma unroll
    for (int g = 0; g < 4; ++g)
      if (g != wn) rmax[r] = fmaxf(rmax[r], redMax[g * 32 + ridx + r]);

  #pragma unroll
  for (int r = 0; r < 8; ++r) {
    float s = 0.0f;
    #pragma unroll
    for (int i = 0; i < 16; ++i) {
      float e = __expf((sacc[i][r] - rmax[r]) * scale);
      sacc[i][r] = e;
      s += e;
    }
    for (int msk = 1; msk < 16; msk <<= 1) s += __shfl_xor(s, msk, 32);
    rsum[r] = s;
  }
  if (ln == 0) {
    #pragma unroll
    for (int r = 0; r < 8; ++r) redSum[wn * 32 + ridx + r] = rsum[r];
  }
  __syncthreads();
  #pragma unroll
  for (int r = 0; r < 8; ++r) {
    #pragma unroll
    for (int g = 0; g < 4; ++g)
      if (g != wn) rsum[r] += redSum[g * 32 + ridx + r];
    rinv[r] = 1.0f / rsum[r];
  }

  // ---- probs out (fp32, nontemporal) + bf16 copy to LDS -------------------
  #pragma unroll
  for (int i = 0; i < 16; ++i) {
    const int lc = i >> 2, nt = i & 3;
    const int lcol = lc * 64 + nt * 16 + (int)ln;        // 0..255
    const int col  = wn * 256 + lcol;                    // global l
    #pragma unroll
    for (int r = 0; r < 8; ++r) {
      const int m = r + 8 * (int)hl;
      float pv = sacc[i][r] * rinv[r];
      __builtin_nontemporal_store(
          pv, &probs[((size_t)bh * AT_T + (t0 + wm * 16 + m)) * AT_T + col]);
      probsLds[wave][m][lcol] = (__bf16)pv;
    }
  }
  __syncthreads();

  // ---- Phase 3: context = probs @ v  (K-dim = l, per wn-quarter) ----------
  v8f cacc[4] = {};
  #pragma unroll
  for (int step = 0; step < 8; ++step) {
    v16bf av = load_a_chunk(&probsLds[wave][ln][32 * step], hl);
    #pragma unroll
    for (int nt = 0; nt < 4; ++nt) {
      v16bf bv = *(const v16bf*)&vT[wn][nt * 16 + (int)ln][32 * step + 16 * (int)hl];
      cacc[nt] = wmma_bf16(av, bv, cacc[nt]);
    }
  }

  // combine the four l-quarters via LDS; wn==0 stores bf16 context
  if (wn > 0) {
    #pragma unroll
    for (int nt = 0; nt < 4; ++nt)
      #pragma unroll
      for (int r = 0; r < 8; ++r)
        ctxPart[wm][wn - 1][r + 8 * (int)hl][nt * 16 + (int)ln] = cacc[nt][r];
  }
  __syncthreads();
  if (wn == 0) {
    #pragma unroll
    for (int nt = 0; nt < 4; ++nt)
      #pragma unroll
      for (int r = 0; r < 8; ++r) {
        const int m = r + 8 * (int)hl;
        float v = cacc[nt][r];
        #pragma unroll
        for (int g = 0; g < 3; ++g) v += ctxPart[wm][g][m][nt * 16 + (int)ln];
        ctx[((size_t)(b * AT_T + t0 + wm * 16 + m)) * AT_DM + hd * 64 + nt * 16 + (int)ln]
            = (__bf16)v;
      }
  }
}

// ---------------------------------------------------------------------------
extern "C" void kernel_launch(void* const* d_in, const int* in_sizes, int n_in,
                              void* d_out, int out_size, void* d_ws, size_t ws_size,
                              hipStream_t stream) {
  (void)in_sizes; (void)n_in; (void)out_size; (void)ws_size;
  const float* x  = (const float*)d_in[0];
  const float* xk = (const float*)d_in[1];
  const float* xv = (const float*)d_in[2];
  const float* pe = (const float*)d_in[3];
  const float* Wq = (const float*)d_in[4];
  const float* bq = (const float*)d_in[5];
  const float* Wk = (const float*)d_in[6];
  const float* bk = (const float*)d_in[7];
  const float* Wv = (const float*)d_in[8];
  const float* bv = (const float*)d_in[9];
  const float* Wp = (const float*)d_in[10];
  const float* Wo = (const float*)d_in[11];
  const float* bo = (const float*)d_in[12];
  const float* pu = (const float*)d_in[13];
  const float* pv = (const float*)d_in[14];

  float* out   = (float*)d_out;                       // [8,1024,512]
  float* probs = out + (size_t)8 * 1024 * 512;        // [8,8,1024,1024]

  __bf16* ws = (__bf16*)d_ws;                         // bf16 intermediates
  __bf16* Qw = ws;                                    // 8192x512
  __bf16* Kw = ws + 4194304;                          // 8192x512
  __bf16* Vw = ws + 8388608;                          // 8192x512
  __bf16* Pw = ws + 12582912;                         // 1024x512
  __bf16* Cw = ws + 13107200;                         // 8192x512 context

  dim3 blk(256);
  dim3 gBig(8192 / GM_TM, 512 / GM_TN);   // (128,4)
  dim3 gPos(1024 / GM_TM, 512 / GM_TN);   // (16,4)

  gemm_bias_kernel<float, __bf16><<<gBig, blk, 0, stream>>>(x,  Wq, bq,      Qw, 8192, 512, 512);
  gemm_bias_kernel<float, __bf16><<<gBig, blk, 0, stream>>>(xk, Wk, bk,      Kw, 8192, 512, 512);
  gemm_bias_kernel<float, __bf16><<<gBig, blk, 0, stream>>>(xv, Wv, bv,      Vw, 8192, 512, 512);
  gemm_bias_kernel<float, __bf16><<<gPos, blk, 0, stream>>>(pe, Wp, nullptr, Pw, 1024, 512, 512);

  const size_t attLds = 256u * 1024u + 1024u;   // 257 KB dynamic LDS (<= 320 KB/WGP)
  (void)hipFuncSetAttribute((const void*)attention_kernel,
                            hipFuncAttributeMaxDynamicSharedMemorySize, (int)attLds);
  attention_kernel<<<dim3(1024 / 32, 64), blk, attLds, stream>>>(
      Qw, Kw, Vw, Pw, pu, pv, probs, Cw);

  gemm_bias_kernel<__bf16, float><<<gBig, blk, 0, stream>>>(Cw, Wo, bo, out, 8192, 512, 512);
}
